// gruaging_rnn_70308614636164
// MI455X (gfx1250) — compile-verified, matrix operand order checked by reading, not run on hardware
//
#include <hip/hip_runtime.h>
#include <hip/hip_bf16.h>

typedef __attribute__((ext_vector_type(16))) __bf16 v16bf;
typedef __attribute__((ext_vector_type(8)))  float  v8f;
typedef int v4i_ __attribute__((vector_size(4 * sizeof(int))));

#define NINPUT 64
#define NHID   256
#define NHA    512
#define NOUT   32
#define BATCH  128
#define SEQ    2048
#define EQI    50
#define FIN    65            // NINPUT + 1 (age channel)
#define KPAD   96            // FIN padded to multiple of 32
#define GDIM   1536          // 3*NHA
#define KCAT   608           // KPAD + NHA
#define NT_G   96            // GDIM/16
#define KT_G   19            // KCAT/32
#define NDP    (NINPUT*NHID + NHID*NHID + NHID)  // 82176
#define NT_A   (NDP/16)      // 5136
#define KT_A   (NHA/32)      // 16

// A-tile swizzle: element (row m in 0..15, k-in-tile ko in 0..31) lives at
// lane l = ((ko>>3)&1)*16 + m, slot j = (ko&7) + ((ko>>4)<<3).
__device__ __forceinline__ int a_lane(int m, int ko) { return (((ko >> 3) & 1) << 4) + m; }
__device__ __forceinline__ int a_slot(int ko)        { return (ko & 7) + ((ko >> 4) << 3); }

#if __has_builtin(__builtin_amdgcn_global_load_async_to_lds_b128)
#define HAVE_ASYNC_LDS 1
#endif

__device__ __forceinline__ void wait_asynccnt0() {
#if __has_builtin(__builtin_amdgcn_s_wait_asynccnt)
  __builtin_amdgcn_s_wait_asynccnt(0);
#else
  asm volatile("s_wait_asynccnt 0" ::: "memory");
#endif
}

__device__ __forceinline__ v8f wmma_bf16(v16bf a, v16bf b, v8f c) {
  return __builtin_amdgcn_wmma_f32_16x16x32_bf16(false, a, false, b,
                                                 (short)0, c, false, false);
}

// ---------------------------------------------------------------------------
// Pack [w_ih | w_hh] (fp32) into bf16 B-operand tiles (32Kx16N), CDNA5 layout:
// tile (nt,kt): lane l holds column n = nt*16+(l&15); element j -> k =
// kt*32 + (l>>4)*8 + (j&7) + (j>=8 ? 16 : 0).  K is [u(96-pad) | h(512)].
// ---------------------------------------------------------------------------
__global__ void k_pack_wgru(const float* __restrict__ wih,
                            const float* __restrict__ whh,
                            __bf16* __restrict__ wp) {
  const long total = (long)NT_G * KT_G * 512;
  for (long idx = (long)blockIdx.x * blockDim.x + threadIdx.x; idx < total;
       idx += (long)gridDim.x * blockDim.x) {
    int e = (int)(idx & 511);
    long tile = idx >> 9;
    int kt = (int)(tile % KT_G);
    int nt = (int)(tile / KT_G);
    int lane = e >> 4, j = e & 15;
    int n = nt * 16 + (lane & 15);
    int k = kt * 32 + ((lane >> 4) << 3) + (j & 7) + ((j >> 3) << 4);
    float v = 0.f;
    if (k < KPAD) { if (k < FIN) v = wih[(long)n * FIN + k]; }
    else           v = whh[(long)n * NHA + (k - KPAD)];
    wp[idx] = (__bf16)v;
  }
}

__global__ void k_pack_wage(const float* __restrict__ wage,
                            __bf16* __restrict__ wp) {
  const long total = (long)NT_A * KT_A * 512;
  for (long idx = (long)blockIdx.x * blockDim.x + threadIdx.x; idx < total;
       idx += (long)gridDim.x * blockDim.x) {
    int e = (int)(idx & 511);
    long tile = idx >> 9;
    int kt = (int)(tile % KT_A);
    int nt = (int)(tile / KT_A);
    int lane = e >> 4, j = e & 15;
    int n = nt * 16 + (lane & 15);
    int k = kt * 32 + ((lane >> 4) << 3) + (j & 7) + ((j >> 3) << 4);
    wp[idx] = (__bf16)wage[(long)n * NHA + k];
  }
}

// ---------------------------------------------------------------------------
// Sequential GRU. 8 workgroups x 16 batches, 8 waves. Per step one fused
// [u_t|h] (16x608) @ Wcat.T (608x1536) bf16 WMMA GEMM. The A operand lives in
// LDS as bf16 pre-swizzled A-tiles (abuf): two ds b128 loads per lane per tile.
// B tiles stream from global (L2-resident, shared by all WGs); an opaque
// memory barrier per step stops LLVM from hoisting them out of the t-loop.
// The 12 n-tiles per wave are split statically: i<8 -> r/z gates (fused
// gi+gh accumulator), i>=8 -> n gates (separate gi_n / gh_n accumulators,
// since the reference needs tanh(i_n + r*h_n)) -- no runtime predicate.
// ---------------------------------------------------------------------------
__global__ __launch_bounds__(256) void k_gru(const float* __restrict__ u,
                                             const float* __restrict__ bih_g,
                                             const float* __restrict__ bhh_g,
                                             const __bf16* __restrict__ wp,
                                             __bf16* __restrict__ hpack) {
  extern __shared__ char smem[];
  __bf16* abuf = (__bf16*)smem;               // 19 tiles x 32 lanes x 16 = 9728
  float* hbuf  = (float*)(abuf + KT_G * 512); // 16 x 512 fp32 h
  float* gacc  = hbuf + 16 * NHA;             // 16 x 1536 fused gate sums
  float* gaccn = gacc + 16 * GDIM;            // 16 x 512 gh_n only
  float* bih   = gaccn + 16 * NHA;            // 1536
  float* bhh   = bih + GDIM;                  // 1536
  const int tid = threadIdx.x, lane = tid & 31, wv = tid >> 5;
  const int mt = blockIdx.x;
  const int b0 = mt * 16;

  for (int i = tid; i < GDIM; i += 256) { bih[i] = bih_g[i]; bhh[i] = bhh_g[i]; }
  for (int i = tid; i < KT_G * 512; i += 256) abuf[i] = (__bf16)0.f;
  for (int i = tid; i < 16 * NHA; i += 256) hbuf[i] = 0.f;
  __syncthreads();

  const int nl = lane & 15, mh = (lane >> 4) << 3;  // D: lane->N, vgpr->M (+8 hi)

  for (int t = 0; t < SEQ; ++t) {
    // opaque barrier: keeps the weight-tile loads inside the time loop
    asm volatile("" ::: "memory");

    // stage u_t (65 channels, zero-padded to 96) straight into A-tile layout
    for (int s = tid; s < 16 * KPAD; s += 256) {
      int m = s / KPAD, f = s - m * KPAD;
      float v = 0.f;
      if (f < FIN) v = u[((long)(b0 + m) * SEQ + t) * FIN + f];
      int kt = f >> 5, ko = f & 31;
      abuf[((kt << 5) + a_lane(m, ko)) * 16 + a_slot(ko)] = (__bf16)v;
    }
    __syncthreads();

    // --- r/z gate tiles: nt = wv + 8i, i < 8; single fused accumulator ---
    #pragma clang loop unroll_count(2)
    for (int i = 0; i < 8; ++i) {
      int nt = wv + (i << 3);
      v8f acc = {};
      const __bf16* bp = wp + ((long)(nt * KT_G) << 9) + (lane << 4);
      for (int kt = 0; kt < KT_G; ++kt) {
        v16bf a  = *(const v16bf*)&abuf[((kt << 5) + lane) << 4];
        v16bf bt = *(const v16bf*)(bp + ((long)kt << 9));
        if (kt + 1 < KT_G) __builtin_prefetch(bp + ((long)(kt + 1) << 9), 0, 3);
        acc = wmma_bf16(a, bt, acc);
      }
      #pragma unroll
      for (int v = 0; v < 8; ++v)
        gacc[(v + mh) * GDIM + nt * 16 + nl] = acc[v];
    }

    // --- n gate tiles: nt = 64 + wv + 8i, i < 4; gi_n (kt<3) | gh_n (kt>=3) ---
    #pragma clang loop unroll_count(2)
    for (int i = 0; i < 4; ++i) {
      int nt = 64 + wv + (i << 3);
      v8f accI = {}; v8f accH = {};
      const __bf16* bp = wp + ((long)(nt * KT_G) << 9) + (lane << 4);
      #pragma unroll
      for (int kt = 0; kt < 3; ++kt) {
        v16bf a  = *(const v16bf*)&abuf[((kt << 5) + lane) << 4];
        v16bf bt = *(const v16bf*)(bp + ((long)kt << 9));
        accI = wmma_bf16(a, bt, accI);
      }
      for (int kt = 3; kt < KT_G; ++kt) {
        v16bf a  = *(const v16bf*)&abuf[((kt << 5) + lane) << 4];
        v16bf bt = *(const v16bf*)(bp + ((long)kt << 9));
        if (kt + 1 < KT_G) __builtin_prefetch(bp + ((long)(kt + 1) << 9), 0, 3);
        accH = wmma_bf16(a, bt, accH);
      }
      #pragma unroll
      for (int v = 0; v < 8; ++v) {
        gacc[(v + mh) * GDIM + nt * 16 + nl]          = accI[v];
        gaccn[(v + mh) * NHA + (nt - 64) * 16 + nl]   = accH[v];
      }
    }
    __syncthreads();

    // elementwise GRU update: thread -> (m, 32 consecutive j); write h both as
    // fp32 (hbuf) and bf16 into the swizzled A-tiles (K offset KPAD+j).
    {
      int m = tid >> 4, j0 = (tid & 15) << 5;
      for (int j = j0; j < j0 + 32; ++j) {
        float xr  = gacc[m * GDIM + j] + bih[j] + bhh[j];
        float xz  = gacc[m * GDIM + NHA + j] + bih[NHA + j] + bhh[NHA + j];
        float xin = gacc[m * GDIM + 2 * NHA + j] + bih[2 * NHA + j];
        float xhn = gaccn[m * NHA + j] + bhh[2 * NHA + j];
        float r = 1.f / (1.f + __expf(-xr));
        float z = 1.f / (1.f + __expf(-xz));
        float nv = tanhf(xin + r * xhn);
        float hn = (1.f - z) * nv + z * hbuf[m * NHA + j];
        hbuf[m * NHA + j] = hn;
        int k = KPAD + j, kt = k >> 5, ko = k & 31;
        abuf[((kt << 5) + a_lane(m, ko)) * 16 + a_slot(ko)] = (__bf16)hn;
      }
    }
    __syncthreads();
  }

  // emit final h as packed bf16 A-panel: hpack[(mt*16+kt)*32 + l][16]
  for (int s = tid; s < 512; s += 256) {
    int kt = s >> 5, l = s & 31;
    int m = l & 15;
    int kb = (kt << 5) + ((l >> 4) << 3);
    #pragma unroll
    for (int j = 0; j < 16; ++j) {
      int k = kb + (j & 7) + ((j >> 3) << 4);
      hpack[(((long)mt * 16 + kt) * 32 + l) * 16 + j] = (__bf16)hbuf[m * NHA + k];
    }
  }
}

// ---------------------------------------------------------------------------
// dparams = h_age (128x512) @ W_age.T (512x82176) + b_age.  WMMA GEMM:
// block = (8 nt) x (1 mt). The bf16 A panel arrives pre-packed from k_gru, so
// LDS staging is a straight 16 KB copy -> gfx1250 async global->LDS loads.
// ---------------------------------------------------------------------------
__global__ __launch_bounds__(256) void k_dparams(const __bf16* __restrict__ hpack,
                                                 const __bf16* __restrict__ wp,
                                                 const float* __restrict__ b_age,
                                                 float* __restrict__ dparams) {
  __shared__ __bf16 apan[16 * 32 * 16];   // [kt][lane][16]
  const int tid = threadIdx.x, lane = tid & 31, wv = tid >> 5;
  const int mt = blockIdx.y;
  const __bf16* src = hpack + (long)mt * (16 * 32 * 16);
#ifdef HAVE_ASYNC_LDS
  #pragma unroll
  for (int i = 0; i < 4; ++i) {           // 1024 x 16B chunks, 4 per thread
    int c = tid + i * 256;
    __builtin_amdgcn_global_load_async_to_lds_b128(
        (v4i_*)(src + c * 8),
        (v4i_*)(apan + c * 8), 0, 0);
  }
  wait_asynccnt0();
#else
  #pragma unroll
  for (int i = 0; i < 4; ++i) {
    int c = tid + i * 256;
    ((uint4*)apan)[c] = ((const uint4*)src)[c];
  }
#endif
  __syncthreads();

  int nt = blockIdx.x * 8 + wv;
  const __bf16* bp = wp + (((long)nt * KT_A) << 9) + (lane << 4);
  v8f acc = {};
  for (int kt = 0; kt < KT_A; ++kt) {
    v16bf a  = *(const v16bf*)&apan[((kt << 5) + lane) << 4];
    v16bf bt = *(const v16bf*)(bp + ((long)kt << 9));
    if (kt + 1 < KT_A) __builtin_prefetch(bp + ((long)(kt + 1) << 9), 0, 3);
    acc = wmma_bf16(a, bt, acc);
  }
  int nl = lane & 15, mh = (lane >> 4) << 3;
  int gn = nt * 16 + nl;
  float ba = b_age[gn];
  #pragma unroll
  for (int v = 0; v < 8; ++v) {
    int gm = mt * 16 + mh + v;
    dparams[(long)gm * NDP + gn] = acc[v] + ba;
  }
}

// ---------------------------------------------------------------------------
// Per-batch: bias0, 50 fixed-point iters, 2048-step scan with fused Bu_t and
// fused output projection. Aw/Bw/W_out live in LDS (bf16), h/bias fp32.
// One workgroup per batch (Aw differs per batch -> WMMA can't amortize).
// ---------------------------------------------------------------------------
__global__ __launch_bounds__(256) void k_scan(const float* __restrict__ u,
                                              const float* __restrict__ dparams,
                                              const float* __restrict__ W_out,
                                              const float* __restrict__ b_out,
                                              float* __restrict__ out) {
  extern __shared__ char smem[];
  __bf16* AwL = (__bf16*)smem;                  // 256*256
  __bf16* BwL = AwL + NHID * NHID;              // 256*64
  __bf16* WoL = BwL + NHID * NINPUT;            // 32*256
  float* h     = (float*)(WoL + NOUT * NHID);
  float* bias0 = h + NHID;
  float* CwL   = bias0 + NHID;
  float* uS    = CwL + NHID;
  float* outp  = uS + NINPUT;                   // 8*32 partials
  const int tid = threadIdx.x;
  const int b = blockIdx.x;
  const float* dp = dparams + (long)b * NDP;

  for (int i = tid; i < NHID * NHID; i += 256) AwL[i] = (__bf16)dp[NHID * NINPUT + i];
  for (int i = tid; i < NHID * NINPUT; i += 256) BwL[i] = (__bf16)dp[i];
  for (int i = tid; i < NOUT * NHID; i += 256) WoL[i] = (__bf16)W_out[i];
  if (tid < NHID)   CwL[tid] = dp[NHID * NINPUT + NHID * NHID + tid];
  if (tid < NINPUT) uS[tid]  = u[(long)b * SEQ * FIN + tid];   // u0, first 64 ch
  __syncthreads();

  // bias0 = Bw @ u0 + Cw ; h0 = 0
  {
    float a = CwL[tid];
    for (int k = 0; k < NINPUT; ++k) a += (float)BwL[tid * NINPUT + k] * uS[k];
    bias0[tid] = a;
    h[tid] = 0.f;
  }
  __syncthreads();

  // 50 fixed-point iterations: h = tanh(Aw h + bias0)
  for (int it = 0; it < EQI; ++it) {
    float a = bias0[tid];
    for (int k = 0; k < NHID; ++k) a += (float)AwL[tid * NHID + k] * h[k];
    __syncthreads();
    h[tid] = tanhf(a);
    __syncthreads();
  }

  // scan: h = tanh(Aw h + Bw u_t + Cw), out_t = W_out h + b_out
  for (int t = 0; t < SEQ; ++t) {
    if (tid < NINPUT) uS[tid] = u[((long)b * SEQ + t) * FIN + tid];
    __syncthreads();
    float a = CwL[tid];
    for (int k = 0; k < NINPUT; ++k) a += (float)BwL[tid * NINPUT + k] * uS[k];
    for (int k = 0; k < NHID; ++k)   a += (float)AwL[tid * NHID + k] * h[k];
    __syncthreads();
    h[tid] = tanhf(a);
    __syncthreads();
    int o = tid & 31, c = tid >> 5;
    float p = 0.f;
    for (int k = c * 32; k < c * 32 + 32; ++k) p += (float)WoL[o * NHID + k] * h[k];
    outp[c * 32 + o] = p;
    __syncthreads();
    if (tid < NOUT) {
      float s = b_out[tid];
      #pragma unroll
      for (int c2 = 0; c2 < 8; ++c2) s += outp[c2 * 32 + tid];
      out[((long)b * SEQ + t) * NOUT + tid] = s;
    }
    __syncthreads();
  }
}

// ---------------------------------------------------------------------------
extern "C" void kernel_launch(void* const* d_in, const int* in_sizes, int n_in,
                              void* d_out, int out_size, void* d_ws, size_t ws_size,
                              hipStream_t stream) {
  const float* u    = (const float*)d_in[0];
  const float* wih  = (const float*)d_in[1];
  const float* whh  = (const float*)d_in[2];
  const float* bih  = (const float*)d_in[3];
  const float* bhh  = (const float*)d_in[4];
  const float* Wage = (const float*)d_in[5];
  const float* bage = (const float*)d_in[6];
  const float* Wout = (const float*)d_in[7];
  const float* bout = (const float*)d_in[8];
  float* out = (float*)d_out;

  char* ws = (char*)d_ws;
  size_t off = 0;
  auto carve = [&](size_t bytes) { size_t p = off; off += (bytes + 255) & ~(size_t)255; return p; };
  __bf16* wgru_p = (__bf16*)(ws + carve((size_t)NT_G * KT_G * 512 * 2));   // ~1.8 MB
  __bf16* wage_p = (__bf16*)(ws + carve((size_t)NT_A * KT_A * 512 * 2));   // ~84 MB
  __bf16* hpack  = (__bf16*)(ws + carve((size_t)BATCH * NHA * 2));         // 128 KB packed
  float* dparams = (float*)(ws + carve((size_t)BATCH * NDP * 4));          // ~42 MB

  k_pack_wgru<<<912, 256, 0, stream>>>(wih, whh, wgru_p);
  k_pack_wage<<<8192, 256, 0, stream>>>(Wage, wage_p);

  size_t smA = (size_t)KT_G * 512 * 2 +
               (size_t)(16 * NHA + 16 * GDIM + 16 * NHA + 2 * GDIM) * 4;   // ~191 KB
  k_gru<<<BATCH / 16, 256, smA, stream>>>(u, bih, bhh, wgru_p, hpack);

  k_dparams<<<dim3(NT_A / 8, BATCH / 16), 256, 0, stream>>>(hpack, wage_p, bage, dparams);

  size_t smC = (size_t)(NHID * NHID + NHID * NINPUT + NOUT * NHID) * 2 +
               (size_t)(3 * NHID + NINPUT + 256) * 4;                      // ~184 KB
  k_scan<<<BATCH, 256, smC, stream>>>(u, dparams, Wout, bout, out);
}
